// GRU_59820304499092
// MI455X (gfx1250) — matrix operand order repaired and checked by reading, not compiled, not run on hardware
//
#include <hip/hip_runtime.h>
#include <hip/hip_bf16.h>
#include <math.h>

// Problem constants (reference: T=512, B=64, E=300, H=512, V=30000, L=5)
#define TT    512
#define BB    64
#define EIN   300
#define EPAD  320      // pad K to multiple of 32 for WMMA 16x16x32
#define HH    512
#define G3    1536     // 3*H
#define VV    30000
#define LL    5

typedef __bf16 bf16_t;
typedef bf16_t v16bf __attribute__((ext_vector_type(16)));
typedef bf16_t v8bf  __attribute__((ext_vector_type(8)));
typedef float  v8f   __attribute__((ext_vector_type(8)));

__device__ __forceinline__ unsigned short f2bf(float f) {
    union { float f; unsigned int u; } v; v.f = f;
    unsigned int u = v.u;
    return (unsigned short)((u + 0x7FFFu + ((u >> 16) & 1u)) >> 16); // RNE
}

// ---- fp32 -> bf16 with optional K padding (weights + embedding table) ----
__global__ void convert_pad_bf16(const float* __restrict__ src,
                                 unsigned short* __restrict__ dst,
                                 int M, int K, int Kp) {
    int i = blockIdx.x * blockDim.x + threadIdx.x;
    if (i >= M * Kp) return;
    int r = i / Kp, k = i - r * Kp;
    dst[i] = (k < K) ? f2bf(src[(size_t)r * K + k]) : (unsigned short)0;
}

__global__ void zero_f32(float* __restrict__ p, int n) {
    int i = blockIdx.x * blockDim.x + threadIdx.x;
    if (i < n) p[i] = 0.f;
}

// ---- bf16 WMMA GEMM: C[M,N] = A[M,K] * W[N,K]^T, f32 accumulate.
// A row-major bf16 (optionally gathered via rowidx), W row-major [N,K] bf16.
// Each wave computes a 16(M) x 64(N) strip: 1 A-frag reused across 4 B-frags.
// Requires: K % 32 == 0, N % 64 == 0, M % 64 == 0 (grid.y covers 64 rows/block).
__global__ __launch_bounds__(128)
void gemm_bf16_wmma(const bf16_t* __restrict__ A, const int* __restrict__ rowidx,
                    const bf16_t* __restrict__ W, float* __restrict__ C,
                    int M, int N, int K, int lda, int ldw, int ldc) {
    const int lane = threadIdx.x & 31;
    const int wave = threadIdx.x >> 5;
    const int mrow = lane & 15;   // M row within tile / N col within tile
    const int kh   = lane >> 4;   // lane-group 0/1
    const int m0 = blockIdx.y * 64 + wave * 16;
    const int n0 = blockIdx.x * 64;
    if (m0 >= M || n0 >= N) return;   // wave-uniform exit, EXEC stays all-ones

    const int row = m0 + mrow;
    const bf16_t* arow = A + (size_t)(rowidx ? rowidx[row] : row) * lda;
    const bf16_t* wb0 = W + (size_t)(n0 +  0 + mrow) * ldw + kh * 16;
    const bf16_t* wb1 = wb0 + (size_t)16 * ldw;
    const bf16_t* wb2 = wb0 + (size_t)32 * ldw;
    const bf16_t* wb3 = wb0 + (size_t)48 * ldw;

    v8f acc0 = {}, acc1 = {}, acc2 = {}, acc3 = {};

    for (int k0 = 0; k0 < K; k0 += 32) {
        // A fragment: lanes 0-15 hold K {0..15}, lanes 16-31 hold {16..31}
        v8bf alo = *(const v8bf*)(arow + k0 + kh * 16);
        v8bf ahi = *(const v8bf*)(arow + k0 + kh * 16 + 8);
        v16bf a;
#pragma unroll
        for (int i = 0; i < 8; ++i) { a[i] = alo[i]; a[i + 8] = ahi[i]; }

        // B fragments: 16 contiguous halves of W[n][k0 + kh*16 ..]
        v16bf b0 = *(const v16bf*)(wb0 + k0);
        v16bf b1 = *(const v16bf*)(wb1 + k0);
        v16bf b2 = *(const v16bf*)(wb2 + k0);
        v16bf b3 = *(const v16bf*)(wb3 + k0);

        acc0 = __builtin_amdgcn_wmma_f32_16x16x32_bf16(false, a, false, b0, (short)0, acc0, false, false);
        acc1 = __builtin_amdgcn_wmma_f32_16x16x32_bf16(false, a, false, b1, (short)0, acc1, false, false);
        acc2 = __builtin_amdgcn_wmma_f32_16x16x32_bf16(false, a, false, b2, (short)0, acc2, false, false);
        acc3 = __builtin_amdgcn_wmma_f32_16x16x32_bf16(false, a, false, b3, (short)0, acc3, false, false);
    }

#pragma unroll
    for (int r = 0; r < 8; ++r) {
        float* crow = C + (size_t)(m0 + r + kh * 8) * ldc + n0 + mrow;
        crow[0]  = acc0[r];
        crow[16] = acc1[r];
        crow[32] = acc2[r];
        crow[48] = acc3[r];
    }
}

// ---- Fused GRU gate update for one timestep.
// xpt,gh: [B, 3H] (bias-free projections). Gate order (r,z,n) per PyTorch.
// Maintains f32 hidden state + bf16 shadow (next step's WMMA A operand).
template <bool WRITE_YS, bool POOL>
__global__ void gru_gate(const float* __restrict__ xpt, const float* __restrict__ gh,
                         const float* __restrict__ bih, const float* __restrict__ bhh,
                         float* __restrict__ hF, unsigned short* __restrict__ hB,
                         unsigned short* __restrict__ ys, float* __restrict__ pooled_t) {
    int i = blockIdx.x * blockDim.x + threadIdx.x;   // [0, B*H)
    if (i >= BB * HH) return;
    int b = i >> 9;            // /H
    int h = i & (HH - 1);
    size_t rb = (size_t)b * G3;
    float xr = xpt[rb + h]          + bih[h];
    float xz = xpt[rb + HH + h]     + bih[HH + h];
    float xn = xpt[rb + 2 * HH + h] + bih[2 * HH + h];
    float hr = gh[rb + h]           + bhh[h];
    float hz = gh[rb + HH + h]      + bhh[HH + h];
    float hn = gh[rb + 2 * HH + h]  + bhh[2 * HH + h];
    float r = 1.f / (1.f + __expf(-(xr + hr)));
    float z = 1.f / (1.f + __expf(-(xz + hz)));
    float n = tanhf(xn + r * hn);
    float hnew = (1.f - z) * n + z * hF[i];
    hF[i] = hnew;
    hB[i] = f2bf(hnew);
    if (WRITE_YS) ys[i] = f2bf(hnew);
    if (POOL) atomicAdd(&pooled_t[h], hnew * (1.f / BB));
}

// ---- final classifier: out[t,l] = pooled[t,:] . fcW[l,:] + fcb[l] ----
__global__ void fc_kernel(const float* __restrict__ pooled, const float* __restrict__ fcW,
                          const float* __restrict__ fcb, float* __restrict__ out) {
    __shared__ float red[256];
    int t = blockIdx.x, tid = threadIdx.x;
    const float* pr = pooled + (size_t)t * HH;
    for (int l = 0; l < LL; ++l) {
        float p = 0.f;
        for (int h = tid; h < HH; h += 256) p += pr[h] * fcW[l * HH + h];
        red[tid] = p; __syncthreads();
        for (int s = 128; s > 0; s >>= 1) { if (tid < s) red[tid] += red[tid + s]; __syncthreads(); }
        if (tid == 0) out[t * LL + l] = red[0] + fcb[l];
        __syncthreads();
    }
}

extern "C" void kernel_launch(void* const* d_in, const int* in_sizes, int n_in,
                              void* d_out, int out_size, void* d_ws, size_t ws_size,
                              hipStream_t stream) {
    (void)in_sizes; (void)n_in; (void)out_size; (void)ws_size;
    const int*   texts = (const int*)d_in[0];     // [T,B] flat = row index t*B+b
    const float* emb   = (const float*)d_in[1];   // [V,E]
    const float* Wih0  = (const float*)d_in[2];   // [3H,E]
    const float* Whh0  = (const float*)d_in[3];   // [3H,H]
    const float* bih0  = (const float*)d_in[4];
    const float* bhh0  = (const float*)d_in[5];
    const float* Wih1  = (const float*)d_in[6];   // [3H,H]
    const float* Whh1  = (const float*)d_in[7];   // [3H,H]
    const float* bih1  = (const float*)d_in[8];
    const float* bhh1  = (const float*)d_in[9];
    const float* fcW   = (const float*)d_in[10];  // [L,H]
    const float* fcb   = (const float*)d_in[11];
    float* out = (float*)d_out;

    // ---- workspace carve-out (256B aligned) ----
    char* ws = (char*)d_ws;
    size_t off = 0;
    auto carve = [&](size_t bytes) -> void* {
        off = (off + 255) & ~(size_t)255;
        void* p = ws + off;
        off += bytes;
        return p;
    };
    unsigned short* embB  = (unsigned short*)carve((size_t)VV * EPAD * 2);   // 19.2 MB
    unsigned short* Wih0B = (unsigned short*)carve((size_t)G3 * EPAD * 2);
    unsigned short* Whh0B = (unsigned short*)carve((size_t)G3 * HH * 2);
    unsigned short* Wih1B = (unsigned short*)carve((size_t)G3 * HH * 2);
    unsigned short* Whh1B = (unsigned short*)carve((size_t)G3 * HH * 2);
    float*          xp    = (float*)carve((size_t)TT * BB * G3 * 4);         // 201 MB
    unsigned short* ys0B  = (unsigned short*)carve((size_t)TT * BB * HH * 2);// 33.5 MB
    float*          gh    = (float*)carve((size_t)BB * G3 * 4);
    float*          hF    = (float*)carve((size_t)BB * HH * 4);
    unsigned short* hB    = (unsigned short*)carve((size_t)BB * HH * 2);
    float*          pooled= (float*)carve((size_t)TT * HH * 4);

    const int M = TT * BB;           // 32768 rows
    dim3 blk(256), gblk(128);

    // 1) weight / embedding conversion to bf16 (pad E 300 -> 320)
    convert_pad_bf16<<<dim3((VV * EPAD + 255) / 256), blk, 0, stream>>>(emb,  embB,  VV, EIN, EPAD);
    convert_pad_bf16<<<dim3((G3 * EPAD + 255) / 256), blk, 0, stream>>>(Wih0, Wih0B, G3, EIN, EPAD);
    convert_pad_bf16<<<dim3((G3 * HH + 255) / 256),   blk, 0, stream>>>(Whh0, Whh0B, G3, HH, HH);
    convert_pad_bf16<<<dim3((G3 * HH + 255) / 256),   blk, 0, stream>>>(Wih1, Wih1B, G3, HH, HH);
    convert_pad_bf16<<<dim3((G3 * HH + 255) / 256),   blk, 0, stream>>>(Whh1, Whh1B, G3, HH, HH);

    // 2) layer-0 input projection, embedding gather fused via rowidx=texts:
    //    xp[t*B+b,:] = emb[texts[t,b],:] . Wih0^T   (M=32768, N=1536, K=320)
    gemm_bf16_wmma<<<dim3(G3 / 64, M / 64), gblk, 0, stream>>>(
        (const bf16_t*)embB, texts, (const bf16_t*)Wih0B, xp, M, G3, EPAD, EPAD, EPAD, G3);

    // 3) layer-0 recurrent scan
    zero_f32<<<dim3((BB * HH + 255) / 256), blk, 0, stream>>>(hF, BB * HH);
    zero_f32<<<dim3((BB * HH / 2 + 255) / 256), blk, 0, stream>>>((float*)hB, BB * HH / 2);
    for (int t = 0; t < TT; ++t) {
        gemm_bf16_wmma<<<dim3(G3 / 64, 1), gblk, 0, stream>>>(
            (const bf16_t*)hB, nullptr, (const bf16_t*)Whh0B, gh, BB, G3, HH, HH, HH, G3);
        gru_gate<true, false><<<dim3((BB * HH + 255) / 256), blk, 0, stream>>>(
            xp + (size_t)t * BB * G3, gh, bih0, bhh0, hF, hB,
            ys0B + (size_t)t * BB * HH, nullptr);
    }

    // 4) layer-1 input projection: xp = ys0 . Wih1^T (M=32768, N=1536, K=512)
    gemm_bf16_wmma<<<dim3(G3 / 64, M / 64), gblk, 0, stream>>>(
        (const bf16_t*)ys0B, nullptr, (const bf16_t*)Wih1B, xp, M, G3, HH, HH, HH, G3);

    // 5) layer-1 recurrent scan, fused batch-mean pooling
    zero_f32<<<dim3((BB * HH + 255) / 256), blk, 0, stream>>>(hF, BB * HH);
    zero_f32<<<dim3((BB * HH / 2 + 255) / 256), blk, 0, stream>>>((float*)hB, BB * HH / 2);
    zero_f32<<<dim3((TT * HH + 255) / 256), blk, 0, stream>>>(pooled, TT * HH);
    for (int t = 0; t < TT; ++t) {
        gemm_bf16_wmma<<<dim3(G3 / 64, 1), gblk, 0, stream>>>(
            (const bf16_t*)hB, nullptr, (const bf16_t*)Whh1B, gh, BB, G3, HH, HH, HH, G3);
        gru_gate<false, true><<<dim3((BB * HH + 255) / 256), blk, 0, stream>>>(
            xp + (size_t)t * BB * G3, gh, bih1, bhh1, hF, hB,
            nullptr, pooled + (size_t)t * HH);
    }

    // 6) classifier
    fc_kernel<<<dim3(TT), blk, 0, stream>>>(pooled, fcW, fcb, out);
}
